// Encoder_60129542967
// MI455X (gfx1250) — matrix operand order repaired and checked by reading, not compiled, not run on hardware
//
#include <hip/hip_runtime.h>
#include <hip/hip_bf16.h>

typedef __attribute__((ext_vector_type(16))) _Float16 v16h;
typedef __attribute__((ext_vector_type(8)))  float    v8f;

#define HEIGHT 64
#define WIDTH  64
#define FCH    64     // F
#define NGATE  256    // 4F
#define BATCH  8
#define TSTEPS 10
#define BHWF   (BATCH*HEIGHT*WIDTH*FCH)

__device__ __forceinline__ float fast_tanh(float x) {
  // tanh(x) = (e^{2x}-1)/(e^{2x}+1), saturate for large |x|
  float xa = fminf(fmaxf(x, -15.0f), 15.0f);
  float ex = __expf(2.0f * xa);
  return (ex - 1.0f) / (ex + 1.0f);
}

// ---------------- x fp32 [B,T,H,W,1] -> f16 [T,B,H,W] ----------------
__global__ void convlstm_convert_x(const float* __restrict__ x, _Float16* __restrict__ x16, int n) {
  int i = blockIdx.x * blockDim.x + threadIdx.x;
  if (i >= n) return;
  const int HW = HEIGHT * WIDTH;
  int b = i / (TSTEPS * HW);
  int r = i - b * (TSTEPS * HW);
  int t = r / HW;
  int p = r - t * HW;
  x16[(t * BATCH + b) * HW + p] = (_Float16)x[i];
}

// ---- pack [Wx;Wh] into f16 chunk-major B: Bpk[chunk][n(256)][kk(32)] ----
// Tap-major K: k = tap*Cpad + c; tap = 3x3 position, c<Cin -> Wx, Cin<=c<Cin+F -> Wh, else 0.
__global__ void convlstm_pack_w(const float* __restrict__ Wx, const float* __restrict__ Wh,
                                _Float16* __restrict__ Bpk, int Cin, int Cpad, int nchunks) {
  int i = blockIdx.x * blockDim.x + threadIdx.x;
  int total = nchunks * NGATE * 32;
  if (i >= total) return;
  int chunk = i >> 13;           // / (256*32)
  int rem   = i & 8191;
  int n     = rem >> 5;
  int kk    = rem & 31;
  int k   = chunk * 32 + kk;
  int tap = k / Cpad;
  int c   = k - tap * Cpad;
  float w = 0.0f;
  if (tap < 9) {
    int ky = tap / 3, kx = tap - ky * 3;
    if (c < Cin)            w = Wx[((ky * 3 + kx) * Cin + c) * NGATE + n];
    else if (c < Cin + FCH) w = Wh[((ky * 3 + kx) * FCH + (c - Cin)) * NGATE + n];
  }
  Bpk[((size_t)chunk * NGATE + n) * 32 + kk] = (_Float16)w;
}

// ---------------- one ConvLSTM timestep (one layer) ----------------
// grid = B*H workgroups (one image row each); 256 threads = 8 wave32.
// gates[b,y,x,n] = sum_k A[x][k]*Bpk[k][n], A = im2col([x_t | h_prev]) tap-major.
__global__ __launch_bounds__(256)
void convlstm_step(const _Float16* __restrict__ Xseq,   // [T][B][H][W][Cin]
                   const _Float16* __restrict__ Hprev,  // [B][H][W][F]
                   const _Float16* __restrict__ Bpk,    // [nchunks][256][32]
                   const float*    __restrict__ bias,   // [256]
                   float*          __restrict__ Cbuf,   // [B][H][W][F]
                   _Float16*       __restrict__ Hcur,   // [B][H][W][F]
                   _Float16*       __restrict__ SeqOut, // [T][B][H][W][F] or null
                   float*          __restrict__ HOut,
                   float*          __restrict__ COut,
                   int t, int Cin, int Cpad, int nchunks, int write_final) {
  extern __shared__ char smem_raw[];
  _Float16* halo  = (_Float16*)smem_raw;   // [3][W][Cpad]  (phase 1-2)
  float*    gates = (float*)smem_raw;      // [64][256]     (phase 3-4, overlapped)

  const int Ctot  = Cin + FCH;
  const int cblks = Cpad >> 5;             // 32-channel blocks per tap
  const int tid  = threadIdx.x;
  const int wave = tid >> 5;
  const int lane = tid & 31;
  const bool hi  = (tid & 16) != 0;
  const int ln16 = lane & 15;

  const int b = blockIdx.x / HEIGHT;
  const int y = blockIdx.x - b * HEIGHT;

  // ---- phase 1: stage 3-row halo of [x_t | h_prev | zero-pad] into LDS ----
  // decode with shifts only: base -> (ky, xx, c%32); inner loop walks 32-ch blocks.
  for (int base = tid; base < 3 * WIDTH * 32; base += 256) {
    int ky = base >> 11;          // /(W*32)
    int r  = base & 2047;
    int xx = r >> 5;
    int cl = r & 31;
    int yy = y + ky - 1;
    bool rowok = (yy >= 0 && yy < HEIGHT);
    _Float16* hdst = &halo[(ky * WIDTH + xx) * Cpad];
    if (rowok) {
      const _Float16* xrow = Xseq + (((size_t)(t * BATCH + b) * HEIGHT + yy) * WIDTH + xx) * Cin;
      const _Float16* hrow = Hprev + (((size_t)b * HEIGHT + yy) * WIDTH + xx) * FCH;
      for (int cb = 0; cb < cblks; ++cb) {
        int c = cl + (cb << 5);
        _Float16 v = (_Float16)0.0f;
        if (c < Cin)       v = xrow[c];
        else if (c < Ctot) v = hrow[c - Cin];
        hdst[c] = v;
      }
    } else {
      for (int cb = 0; cb < cblks; ++cb) hdst[cl + (cb << 5)] = (_Float16)0.0f;
    }
  }
  __syncthreads();

  // ---- phase 2: implicit GEMM, wave tile = 16 rows x 128 gate cols ----
  const int ms  = (wave & 3) * 16;   // m sub-tile origin (x coordinate)
  const int ng0 = (wave >> 2) * 8;   // first of 8 n-tiles
  const int x0  = ms + ln16;         // this lane's A row (spatial x)
  const int kk  = hi ? 16 : 0;       // B-fragment K offset within chunk
  const int aoff = hi ? 16 : 0;      // A-fragment K offset within chunk

  v8f acc[8];
  #pragma unroll
  for (int i = 0; i < 8; ++i) { v8f z = {}; acc[i] = z; }

  union AF { v16h v; uint4 u[2]; };
  union BF { v16h v; uint4 u[2]; };

  int tap = 0, cblk = 0;
  for (int chunk = 0; chunk < nchunks; ++chunk) {
    // chunk lies entirely inside one tap: (ky,kx) uniform, channels contiguous.
    int ky = tap / 3;
    int kx = tap - ky * 3;
    int xx = x0 + kx - 1;
    AF a;
    if (xx >= 0 && xx < WIDTH) {
      // lane holds A[m=x0][k]: runs k=kb..kb+15 contiguous -> two 16B LDS loads
      const uint4* hp4 = (const uint4*)&halo[(ky * WIDTH + xx) * Cpad + (cblk << 5) + aoff];
      a.u[0] = hp4[0];
      a.u[1] = hp4[1];
    } else {
      a.u[0] = make_uint4(0u, 0u, 0u, 0u);
      a.u[1] = make_uint4(0u, 0u, 0u, 0u);
    }

    const _Float16* bbase = Bpk + (size_t)chunk * NGATE * 32;
    if (chunk + 1 < nchunks)
      __builtin_prefetch(bbase + (size_t)NGATE * 32 + ((ng0 * 16 + ln16) * 32 + kk), 0, 1);

    #pragma unroll
    for (int nt = 0; nt < 8; ++nt) {
      int n = (ng0 + nt) * 16 + ln16;
      BF bf;
      const uint4* bp = (const uint4*)(bbase + n * 32 + kk);  // 32B aligned
      bf.u[0] = bp[0];
      bf.u[1] = bp[1];
      acc[nt] = __builtin_amdgcn_wmma_f32_16x16x32_f16(
          false, a.v, false, bf.v, (short)0, acc[nt], false, false);
    }

    if (++cblk == cblks) { cblk = 0; ++tap; }
  }
  __syncthreads();   // all waves done reading halo; reuse LDS for gates

  // ---- phase 3: accumulators -> gates LDS [64][256] f32 ----
  #pragma unroll
  for (int nt = 0; nt < 8; ++nt) {
    int n = (ng0 + nt) * 16 + ln16;
    #pragma unroll
    for (int r = 0; r < 8; ++r) {
      int m = ms + (hi ? 8 : 0) + r;
      gates[m * NGATE + n] = acc[nt][r];
    }
  }
  __syncthreads();

  // ---- phase 4: elementwise LSTM update ----
  const size_t base = ((size_t)b * HEIGHT + y) * WIDTH;
  for (int e = tid; e < WIDTH * FCH; e += 256) {
    int m  = e >> 6;    // x
    int f0 = e & 63;    // feature
    float gi = gates[m * NGATE + f0]        + bias[f0];
    float gf = gates[m * NGATE + 64 + f0]   + bias[64 + f0];
    float gg = gates[m * NGATE + 128 + f0]  + bias[128 + f0];
    float go = gates[m * NGATE + 192 + f0]  + bias[192 + f0];
    float i_ = fminf(fmaxf(0.2f * gi + 0.5f, 0.0f), 1.0f);
    float f_ = fminf(fmaxf(0.2f * gf + 0.5f, 0.0f), 1.0f);
    float g_ = fast_tanh(gg);
    float o_ = fminf(fmaxf(0.2f * go + 0.5f, 0.0f), 1.0f);
    size_t idx = (base + m) * FCH + f0;
    float c_ = f_ * Cbuf[idx] + i_ * g_;
    float h_ = o_ * fast_tanh(c_);
    Cbuf[idx] = c_;
    Hcur[idx] = (_Float16)h_;
    if (SeqOut) SeqOut[(size_t)t * BHWF + idx] = (_Float16)h_;
    if (write_final) { HOut[idx] = h_; COut[idx] = c_; }
  }
}

extern "C" void kernel_launch(void* const* d_in, const int* in_sizes, int n_in,
                              void* d_out, int out_size, void* d_ws, size_t ws_size,
                              hipStream_t stream) {
  const float* x   = (const float*)d_in[0];
  const float* Wx0 = (const float*)d_in[1];
  const float* Wh0 = (const float*)d_in[2];
  const float* b0  = (const float*)d_in[3];
  const float* Wx1 = (const float*)d_in[4];
  const float* Wh1 = (const float*)d_in[5];
  const float* b1  = (const float*)d_in[6];
  float* out = (float*)d_out;

  // Tap-major K, per-tap channels padded to multiple of 32.
  const int CP0 = 96,  NC0 = 9 * CP0 / 32;   // layer0: 65 -> 96, 27 chunks
  const int CP1 = 128, NC1 = 9 * CP1 / 32;   // layer1: 128, 36 chunks

  // ---- workspace partition (bump allocator, 256B aligned) ----
  size_t off = 0;
  auto alloc = [&](size_t bytes) -> void* {
    void* p = (char*)d_ws + off;
    off += (bytes + 255) & ~(size_t)255;
    return p;
  };
  _Float16* x16  = (_Float16*)alloc((size_t)TSTEPS * BATCH * HEIGHT * WIDTH * 2);
  _Float16* out0 = (_Float16*)alloc((size_t)TSTEPS * BHWF * 2);
  _Float16* hA   = (_Float16*)alloc((size_t)BHWF * 2);
  _Float16* hB   = (_Float16*)alloc((size_t)BHWF * 2);
  float*    cbuf = (float*)   alloc((size_t)BHWF * 4);
  _Float16* Bpk0 = (_Float16*)alloc((size_t)NC0 * NGATE * 32 * 2);
  _Float16* Bpk1 = (_Float16*)alloc((size_t)NC1 * NGATE * 32 * 2);
  (void)ws_size; (void)in_sizes; (void)n_in; (void)out_size;

  // ---- prep ----
  {
    int n = BATCH * TSTEPS * HEIGHT * WIDTH;
    convlstm_convert_x<<<(n + 255) / 256, 256, 0, stream>>>(x, x16, n);
  }
  {
    int n0 = NC0 * NGATE * 32;
    convlstm_pack_w<<<(n0 + 255) / 256, 256, 0, stream>>>(Wx0, Wh0, Bpk0, 1, CP0, NC0);
    int n1 = NC1 * NGATE * 32;
    convlstm_pack_w<<<(n1 + 255) / 256, 256, 0, stream>>>(Wx1, Wh1, Bpk1, 64, CP1, NC1);
  }

  const size_t shmem = (size_t)64 * NGATE * 4;  // 64 KB: max(halo = 3*64*Cpad*2, gates)
  const int grid = BATCH * HEIGHT;

  // ---- layer 0 ----
  hipMemsetAsync(hA, 0, (size_t)BHWF * 2, stream);
  hipMemsetAsync(cbuf, 0, (size_t)BHWF * 4, stream);
  {
    _Float16 *hp = hA, *hc = hB;
    for (int t = 0; t < TSTEPS; ++t) {
      convlstm_step<<<grid, 256, shmem, stream>>>(
          x16, hp, Bpk0, b0, cbuf, hc, out0,
          out + 0 * (size_t)BHWF, out + 1 * (size_t)BHWF,
          t, 1, CP0, NC0, (t == TSTEPS - 1) ? 1 : 0);
      _Float16* tmp = hp; hp = hc; hc = tmp;
    }
  }

  // ---- layer 1 ----
  hipMemsetAsync(hA, 0, (size_t)BHWF * 2, stream);
  hipMemsetAsync(cbuf, 0, (size_t)BHWF * 4, stream);
  {
    _Float16 *hp = hA, *hc = hB;
    for (int t = 0; t < TSTEPS; ++t) {
      convlstm_step<<<grid, 256, shmem, stream>>>(
          out0, hp, Bpk1, b1, cbuf, hc, (_Float16*)nullptr,
          out + 2 * (size_t)BHWF, out + 3 * (size_t)BHWF,
          t, 64, CP1, NC1, (t == TSTEPS - 1) ? 1 : 0);
      _Float16* tmp = hp; hp = hc; hc = tmp;
    }
  }
}
